// MultiHeadLatentAttention_11536282157171
// MI455X (gfx1250) — matrix operand order repaired and checked by reading, not compiled, not run on hardware
//
#include <hip/hip_runtime.h>
#include <math.h>

// ---------------------------------------------------------------------------
// Types for CDNA5 WMMA (wave32): v_wmma_f32_16x16x32_bf16
// ---------------------------------------------------------------------------
typedef __attribute__((ext_vector_type(8)))  float  v8f;
typedef __attribute__((ext_vector_type(4)))  __bf16 bf16x4;
typedef __attribute__((ext_vector_type(8)))  __bf16 bf16x8;
typedef __attribute__((ext_vector_type(16))) __bf16 bf16x16;

union Frag16 { bf16x16 v; bf16x8 h[2]; };

__device__ __forceinline__ v8f wmma_bf16(const Frag16& a, const Frag16& b, v8f c) {
  // (neg_a, A, neg_b, B, c_mod, C, reuse_a, reuse_b)
  return __builtin_amdgcn_wmma_f32_16x16x32_bf16(false, a.v, false, b.v,
                                                 (short)0, c, false, false);
}

__device__ __forceinline__ v8f v8f_zero() {
  v8f z = {0.f,0.f,0.f,0.f,0.f,0.f,0.f,0.f};
  return z;
}

// ---------------------------------------------------------------------------
// Generic GEMM: Out[M,N] = X[M,K] @ W[N,K]^T   (fp32 in/out, bf16 WMMA core)
// Block: 256 thr (8 waves). Block tile 128x64, wave tile 32x32 (2x2 WMMA).
// Double-buffered LDS: next tile's global loads overlap current tile's math.
// M = gridDim.y*128, N = gridDim.x*64.  K % 32 == 0.
// ---------------------------------------------------------------------------
__device__ __forceinline__ void gemm_load_tile(
    const float* __restrict__ X, const float* __restrict__ W,
    int m0, int n0, int K, int k0, int tid,
    float4 (&av)[4], float4 (&bv)[2])
{
#pragma unroll
  for (int i = 0; i < 4; ++i) {
    int f = tid + i * 256;
    int row = f >> 3, c4 = (f & 7) * 4;
    av[i] = *(const float4*)(X + (size_t)(m0 + row) * K + k0 + c4);
  }
#pragma unroll
  for (int i = 0; i < 2; ++i) {
    int f = tid + i * 256;
    int row = f >> 3, c4 = (f & 7) * 4;
    bv[i] = *(const float4*)(W + (size_t)(n0 + row) * K + k0 + c4);
  }
}

__device__ __forceinline__ void gemm_store_tile(
    __bf16* shA, __bf16* shB, int tid,
    const float4 (&av)[4], const float4 (&bv)[2])
{
#pragma unroll
  for (int i = 0; i < 4; ++i) {
    int f = tid + i * 256;
    int row = f >> 3, c4 = (f & 7) * 4;
    bf16x4 b;
    b[0] = (__bf16)av[i].x; b[1] = (__bf16)av[i].y;
    b[2] = (__bf16)av[i].z; b[3] = (__bf16)av[i].w;
    *(bf16x4*)(shA + row * 40 + c4) = b;
  }
#pragma unroll
  for (int i = 0; i < 2; ++i) {
    int f = tid + i * 256;
    int row = f >> 3, c4 = (f & 7) * 4;
    bf16x4 b;
    b[0] = (__bf16)bv[i].x; b[1] = (__bf16)bv[i].y;
    b[2] = (__bf16)bv[i].z; b[3] = (__bf16)bv[i].w;
    *(bf16x4*)(shB + row * 40 + c4) = b;
  }
}

__global__ __launch_bounds__(256) void gemm_xwT_kernel(
    const float* __restrict__ X, const float* __restrict__ W,
    float* __restrict__ Out, int N, int K)
{
  __shared__ __bf16 shA[2][128 * 40];   // row stride 40 (80B, 16B aligned)
  __shared__ __bf16 shB[2][64 * 40];

  const int tid  = threadIdx.x;
  const int wave = tid >> 5, lane = tid & 31;
  const int lo = lane & 15, hi = lane >> 4;
  const int wr = wave >> 1, wc = wave & 1;     // 4x2 wave grid
  const int m0 = blockIdx.y * 128, n0 = blockIdx.x * 64;

  v8f acc00 = v8f_zero(), acc01 = v8f_zero();
  v8f acc10 = v8f_zero(), acc11 = v8f_zero();

  const int nt = K >> 5;
  float4 av[4];
  float4 bv[2];

  // prologue: tile 0 -> regs -> LDS buffer 0
  gemm_load_tile(X, W, m0, n0, K, 0, tid, av, bv);
  gemm_store_tile(shA[0], shB[0], tid, av, bv);

  for (int it = 0; it < nt; ++it) {
    __syncthreads();                       // buffer (it&1) visible; prev reads done
    const int cur = it & 1;

    // issue next tile's global loads early (overlap with WMMA below)
    if (it + 1 < nt)
      gemm_load_tile(X, W, m0, n0, K, (it + 1) << 5, tid, av, bv);
    // prefetch tile after next (global_prefetch_b8)
    if (it + 2 < nt) {
      __builtin_prefetch(X + (size_t)(m0 + (tid >> 1)) * K + ((it + 2) << 5), 0, 1);
      if (tid < 64)
        __builtin_prefetch(W + (size_t)(n0 + tid) * K + ((it + 2) << 5), 0, 1);
    }

    // A fragments (16x32): row lo, contiguous 8 at 8*hi and 16+8*hi
    Frag16 a0, a1, b0, b1;
    const __bf16* ar0 = shA[cur] + (wr * 32 + lo) * 40;
    const __bf16* ar1 = shA[cur] + (wr * 32 + 16 + lo) * 40;
    a0.h[0] = *(const bf16x8*)(ar0 + 8 * hi);
    a0.h[1] = *(const bf16x8*)(ar0 + 16 + 8 * hi);
    a1.h[0] = *(const bf16x8*)(ar1 + 8 * hi);
    a1.h[1] = *(const bf16x8*)(ar1 + 16 + 8 * hi);
    // B fragments (32x16 from B^T rows): row lo, contiguous 8 at 8*hi and 16+8*hi
    const __bf16* br0 = shB[cur] + (wc * 32 + lo) * 40;
    const __bf16* br1 = shB[cur] + (wc * 32 + 16 + lo) * 40;
    b0.h[0] = *(const bf16x8*)(br0 + 8 * hi);
    b0.h[1] = *(const bf16x8*)(br0 + 16 + 8 * hi);
    b1.h[0] = *(const bf16x8*)(br1 + 8 * hi);
    b1.h[1] = *(const bf16x8*)(br1 + 16 + 8 * hi);

    acc00 = wmma_bf16(a0, b0, acc00);
    acc01 = wmma_bf16(a0, b1, acc01);
    acc10 = wmma_bf16(a1, b0, acc10);
    acc11 = wmma_bf16(a1, b1, acc11);

    // stage next tile into the other buffer (readers of it finished pre-barrier)
    if (it + 1 < nt)
      gemm_store_tile(shA[1 - cur], shB[1 - cur], tid, av, bv);
  }

  // C/D layout: row r + 8*hi, col lo
#pragma unroll
  for (int r = 0; r < 8; ++r) {
    int mr0 = m0 + wr * 32 + r + 8 * hi;
    int mr1 = mr0 + 16;
    int nc0 = n0 + wc * 32 + lo;
    int nc1 = nc0 + 16;
    Out[(size_t)mr0 * N + nc0] = acc00[r];
    Out[(size_t)mr0 * N + nc1] = acc01[r];
    Out[(size_t)mr1 * N + nc0] = acc10[r];
    Out[(size_t)mr1 * N + nc1] = acc11[r];
  }
}

// ---------------------------------------------------------------------------
// RMS norm over last dim (one row per block, 256 threads)
// ---------------------------------------------------------------------------
template <int LEN>
__global__ __launch_bounds__(256) void rmsnorm_kernel(
    const float* __restrict__ in, const float* __restrict__ w,
    float* __restrict__ out, int in_stride, int out_stride)
{
  __shared__ float red[256];
  const int tid = threadIdx.x;
  constexpr int NV = LEN / 256;
  const float* src = in + (size_t)blockIdx.x * in_stride;
  float v[NV];
  float ss = 0.f;
#pragma unroll
  for (int i = 0; i < NV; ++i) { v[i] = src[tid + (i << 8)]; ss += v[i] * v[i]; }
  red[tid] = ss;
  __syncthreads();
  for (int s = 128; s > 0; s >>= 1) {
    if (tid < s) red[tid] += red[tid + s];
    __syncthreads();
  }
  float scale = rsqrtf(red[0] / (float)LEN + 1e-6f);
  float* dst = out + (size_t)blockIdx.x * out_stride;
#pragma unroll
  for (int i = 0; i < NV; ++i)
    dst[tid + (i << 8)] = v[i] * scale * w[tid + (i << 8)];
}

// ---------------------------------------------------------------------------
// RoPE + pack to bf16 head-major layouts: q/k/v -> [b][h][t][128]
// qbuf: [m][h*128 + j] (j<64 nope, j>=64 rope), kvbuf: [m][h*192 + j]
// kvabuf: [m][576], rope part at cols 512..575 (shared across heads)
// Block 128 (d index), grid 4096 (m = b*T + t)
// ---------------------------------------------------------------------------
__global__ __launch_bounds__(128) void rope_pack_kernel(
    const float* __restrict__ qbuf, const float* __restrict__ kvbuf,
    const float* __restrict__ kvabuf,
    __bf16* __restrict__ qout, __bf16* __restrict__ kout,
    __bf16* __restrict__ vout)
{
  const int m = blockIdx.x;
  const int t = m & 2047, b = m >> 11;
  const int d = threadIdx.x;

  // v: kv[..., 64:192]
#pragma unroll
  for (int h = 0; h < 16; ++h)
    vout[(((size_t)(b * 16 + h) * 2048) + t) * 128 + d] =
        (__bf16)kvbuf[(size_t)m * 3072 + h * 192 + 64 + d];

  if (d < 64) {
#pragma unroll
    for (int h = 0; h < 16; ++h) {
      size_t o = (((size_t)(b * 16 + h) * 2048) + t) * 128 + d;
      qout[o] = (__bf16)qbuf[(size_t)m * 2048 + h * 128 + d];
      kout[o] = (__bf16)kvbuf[(size_t)m * 3072 + h * 192 + d];
    }
  } else {
    const int i = (d - 64) >> 1;
    const int odd = d & 1;
    // inv = theta^(-i/32) ; theta = 10000 -> ln = 9.210340371976184
    float inv = __expf(-(float)i * (9.210340371976184f / 32.0f));
    float ang = (float)t * inv;
    float c = cosf(ang), s = sinf(ang);
    float kxr = kvabuf[(size_t)m * 576 + 512 + 2 * i];
    float kxi = kvabuf[(size_t)m * 576 + 512 + 2 * i + 1];
    __bf16 kb = (__bf16)(odd ? (kxr * s + kxi * c) : (kxr * c - kxi * s));
#pragma unroll
    for (int h = 0; h < 16; ++h) {
      float qxr = qbuf[(size_t)m * 2048 + h * 128 + 64 + 2 * i];
      float qxi = qbuf[(size_t)m * 2048 + h * 128 + 64 + 2 * i + 1];
      float qv = odd ? (qxr * s + qxi * c) : (qxr * c - qxi * s);
      size_t o = (((size_t)(b * 16 + h) * 2048) + t) * 128 + d;
      qout[o] = (__bf16)qv;
      kout[o] = kb;
    }
  }
}

// ---------------------------------------------------------------------------
// Flash attention, causal, d=128, scale=1/8.
// Block 128 thr (4 waves); each wave owns 16 q rows; block q tile = 64.
// Key tiles of 32: 8 QK^T WMMAs + 8 PV WMMAs per tile.
// K/V tiles register-staged: global loads for tile it+1 are issued right
// after tile it becomes visible, overlapping the S/softmax/PV math.
// Q/K/V bf16 [b][h][t][128] -> Out fp32 [b][t][h*128+d]
// ---------------------------------------------------------------------------
__global__ __launch_bounds__(128) void mla_attn_kernel(
    const __bf16* __restrict__ Q, const __bf16* __restrict__ Kg,
    const __bf16* __restrict__ Vg, float* __restrict__ Out)
{
  __shared__ __bf16 shK[32 * 136];      // [key][d], stride 136 (272B)
  __shared__ __bf16 shV[128 * 40];      // transposed [d][key], stride 40
  __shared__ __bf16 shP[4 * 16 * 40];   // per-wave P staging [m][key], stride 40

  const int tid  = threadIdx.x;
  const int wave = tid >> 5, lane = tid & 31;
  const int lo = lane & 15, hi = lane >> 4;
  const int qtile = blockIdx.x, h = blockIdx.y, b = blockIdx.z;
  const int qbase = qtile * 64 + wave * 16;
  const size_t head = (size_t)(b * 16 + h) * 2048;

  // Q fragments: 16 rows x 128d as 4 A-fragments of 16x32
  Frag16 qa[4];
  {
    const __bf16* qrow = Q + (head + qbase + lo) * 128;
#pragma unroll
    for (int c = 0; c < 4; ++c) {
      qa[c].h[0] = *(const bf16x8*)(qrow + c * 32 + 8 * hi);
      qa[c].h[1] = *(const bf16x8*)(qrow + c * 32 + 16 + 8 * hi);
    }
  }

  v8f oacc[8];
#pragma unroll
  for (int n = 0; n < 8; ++n) oacc[n] = v8f_zero();
  float mrow[8], lrow[8];
#pragma unroll
  for (int r = 0; r < 8; ++r) { mrow[r] = -3.0e38f; lrow[r] = 0.f; }

  const int nt = (qtile * 64 + 64) >> 5;   // causal: key tiles needed
  bf16x8 kreg[4], vreg[4];

  // prologue: tile 0 -> regs
#pragma unroll
  for (int i = 0; i < 4; ++i) {
    int f = tid + i * 128;
    int row = f >> 4, col = (f & 15) * 8;
    kreg[i] = *(const bf16x8*)(Kg + (head + row) * 128 + col);
    vreg[i] = *(const bf16x8*)(Vg + (head + row) * 128 + col);
  }

  for (int it = 0; it < nt; ++it) {
    const int kt = it << 5;
    // stage regs -> LDS (K direct, V transposed [d][key])
#pragma unroll
    for (int i = 0; i < 4; ++i) {
      int f = tid + i * 128;
      int row = f >> 4, col = (f & 15) * 8;
      *(bf16x8*)(shK + row * 136 + col) = kreg[i];
#pragma unroll
      for (int j = 0; j < 8; ++j) shV[(col + j) * 40 + row] = vreg[i][j];
    }
    __syncthreads();

    // issue next tile's global loads early
    if (it + 1 < nt) {
#pragma unroll
      for (int i = 0; i < 4; ++i) {
        int f = tid + i * 128;
        int row = f >> 4, col = (f & 15) * 8;
        kreg[i] = *(const bf16x8*)(Kg + (head + kt + 32 + row) * 128 + col);
        vreg[i] = *(const bf16x8*)(Vg + (head + kt + 32 + row) * 128 + col);
      }
    }

    // S = Q K^T for two 16-key sub-tiles
    float s[2][8];
#pragma unroll
    for (int j = 0; j < 2; ++j) {
      v8f sa = v8f_zero();
      const __bf16* krow = shK + (16 * j + lo) * 136;
#pragma unroll
      for (int c = 0; c < 4; ++c) {
        Frag16 kb;
        kb.h[0] = *(const bf16x8*)(krow + c * 32 + 8 * hi);
        kb.h[1] = *(const bf16x8*)(krow + c * 32 + 16 + 8 * hi);
        sa = wmma_bf16(qa[c], kb, sa);
      }
#pragma unroll
      for (int r = 0; r < 8; ++r) {
        int qg = qbase + r + 8 * hi;
        int kg = kt + 16 * j + lo;
        float v = sa[r] * 0.125f;                 // ROPE^-0.5 = 1/8
        s[j][r] = (kg > qg) ? -1.0e30f : v;       // causal mask
      }
    }

    // online softmax (row stats reduced across 16-lane halves)
#pragma unroll
    for (int r = 0; r < 8; ++r) {
      float tm = fmaxf(s[0][r], s[1][r]);
#pragma unroll
      for (int msk = 1; msk < 16; msk <<= 1)
        tm = fmaxf(tm, __shfl_xor(tm, msk, 32));
      float mn = fmaxf(mrow[r], tm);
      float alpha = __expf(mrow[r] - mn);
      mrow[r] = mn;
      float p0 = __expf(s[0][r] - mn);
      float p1 = __expf(s[1][r] - mn);
      float ps = p0 + p1;
#pragma unroll
      for (int msk = 1; msk < 16; msk <<= 1)
        ps += __shfl_xor(ps, msk, 32);
      lrow[r] = lrow[r] * alpha + ps;
#pragma unroll
      for (int n = 0; n < 8; ++n) oacc[n][r] *= alpha;
      // stage P (C-layout -> LDS row-major 16x32 per wave)
      __bf16* prow = shP + wave * 16 * 40 + (r + 8 * hi) * 40;
      prow[lo]      = (__bf16)p0;
      prow[16 + lo] = (__bf16)p1;
    }
    __syncthreads();

    // O += P V : P as A fragment (16x32), V^T rows as B fragments
    Frag16 pa;
    const __bf16* pr = shP + wave * 16 * 40 + lo * 40;
    pa.h[0] = *(const bf16x8*)(pr + 8 * hi);
    pa.h[1] = *(const bf16x8*)(pr + 16 + 8 * hi);
#pragma unroll
    for (int n = 0; n < 8; ++n) {
      Frag16 vb;
      const __bf16* vr = shV + (n * 16 + lo) * 40;
      vb.h[0] = *(const bf16x8*)(vr + 8 * hi);
      vb.h[1] = *(const bf16x8*)(vr + 16 + 8 * hi);
      oacc[n] = wmma_bf16(pa, vb, oacc[n]);
    }
    __syncthreads();   // everyone done with shK/shV before next staging
  }

  // normalize and write [b][t][h*128+d] fp32
#pragma unroll
  for (int r = 0; r < 8; ++r) {
    float inv_l = 1.0f / lrow[r];
    int qg = qbase + r + 8 * hi;
    float* orow = Out + ((size_t)(b * 2048 + qg)) * 2048 + h * 128;
#pragma unroll
    for (int n = 0; n < 8; ++n)
      orow[n * 16 + lo] = oacc[n][r] * inv_l;
  }
}

// ---------------------------------------------------------------------------
// Host-side pipeline
// ---------------------------------------------------------------------------
extern "C" void kernel_launch(void* const* d_in, const int* in_sizes, int n_in,
                              void* d_out, int out_size, void* d_ws, size_t ws_size,
                              hipStream_t stream)
{
  (void)in_sizes; (void)n_in; (void)out_size; (void)ws_size;
  const float* x      = (const float*)d_in[0];
  // d_in[1] = mask (causal handled analytically)
  const float* w_qa   = (const float*)d_in[2];
  const float* qa_nw  = (const float*)d_in[3];
  const float* w_qb   = (const float*)d_in[4];
  const float* w_kva  = (const float*)d_in[5];
  const float* kva_nw = (const float*)d_in[6];
  const float* w_kvb  = (const float*)d_in[7];
  const float* w_o    = (const float*)d_in[8];
  float* out = (float*)d_out;

  char* p = (char*)d_ws;
  float*  qc    = (float*)p;  p += (size_t)4096 * 1536 * sizeof(float);
  float*  qfull = (float*)p;  p += (size_t)4096 * 2048 * sizeof(float);
  float*  kva   = (float*)p;  p += (size_t)4096 * 576  * sizeof(float);
  float*  kvc   = (float*)p;  p += (size_t)4096 * 512  * sizeof(float);
  float*  kv    = (float*)p;  p += (size_t)4096 * 3072 * sizeof(float);
  __bf16* qbf   = (__bf16*)p; p += (size_t)4096 * 2048 * sizeof(__bf16);
  __bf16* kbf   = (__bf16*)p; p += (size_t)4096 * 2048 * sizeof(__bf16);
  __bf16* vbf   = (__bf16*)p; p += (size_t)4096 * 2048 * sizeof(__bf16);
  float*  att   = (float*)p;  p += (size_t)4096 * 2048 * sizeof(float);

  // 1) q_c = x @ w_qa^T               (4096x2048 @ 2048x1536)
  gemm_xwT_kernel<<<dim3(1536 / 64, 4096 / 128), 256, 0, stream>>>(x, w_qa, qc, 1536, 2048);
  // 2) rms_norm(q_c) in-place
  rmsnorm_kernel<1536><<<4096, 256, 0, stream>>>(qc, qa_nw, qc, 1536, 1536);
  // 3) q = q_c @ w_qb^T               (-> 4096x2048)
  gemm_xwT_kernel<<<dim3(2048 / 64, 32), 256, 0, stream>>>(qc, w_qb, qfull, 2048, 1536);
  // 4) kv_a = x @ w_kva^T             (-> 4096x576)
  gemm_xwT_kernel<<<dim3(576 / 64, 32), 256, 0, stream>>>(x, w_kva, kva, 576, 2048);
  // 5) kv_c = rms_norm(kv_a[:,:512])
  rmsnorm_kernel<512><<<4096, 256, 0, stream>>>(kva, kva_nw, kvc, 576, 512);
  // 6) kv = kv_c @ w_kvb^T            (-> 4096x3072)
  gemm_xwT_kernel<<<dim3(3072 / 64, 32), 256, 0, stream>>>(kvc, w_kvb, kv, 3072, 512);
  // 7) RoPE + pack to bf16 head-major q/k/v
  rope_pack_kernel<<<4096, 128, 0, stream>>>(qfull, kv, kva, qbf, kbf, vbf);
  // 8) causal flash attention -> att (fp32, [b][t][h*128+d])
  mla_attn_kernel<<<dim3(32, 16, 2), 128, 0, stream>>>(qbf, kbf, vbf, att);
  // 9) out = att @ w_o^T              (-> 4096x2048)
  gemm_xwT_kernel<<<dim3(32, 32), 256, 0, stream>>>(att, w_o, out, 2048, 2048);
}
